// DividedSpaceTimeAttention_13116830121995
// MI455X (gfx1250) — compile-verified, matrix operand order checked by reading, not run on hardware
//
#include <hip/hip_runtime.h>

typedef __attribute__((ext_vector_type(16))) __bf16 v16bf;
typedef __attribute__((ext_vector_type(8)))  __bf16 v8bf;
typedef __attribute__((ext_vector_type(8)))  float  v8f;
typedef __attribute__((ext_vector_type(4)))  float  v4f;
typedef __attribute__((ext_vector_type(4)))  unsigned int u32x4;
typedef __attribute__((ext_vector_type(8)))  int i32x8;
typedef __attribute__((ext_vector_type(4)))  int i32x4;

#define B_  8
#define T_  8
#define NS  196          // spatial tokens per frame
#define D_  768
#define H_  12
#define HD  64
#define TN  (T_*NS)      // 1568
#define M_  (B_*TN)      // 12544 rows
#define QKN (3*D_)       // 2304 qkv columns

// ---------------------------------------------------------------------------
// Tensor Data Mover: 2D tile (tile0 x tile1, bf16) global -> LDS, with LDS row
// padding. Descriptor per CDNA5 ISA:
//   group0: count=1 | lds_addr | global_addr[56:0] | type=2
//   group1: data_size=1(2B), pad_enable, pad_interval code, pad_amount code,
//           tensor_dim0/1, tile_dim0/1, tensor_dim0_stride
// Issued once per wave (EXEC ignored); tracked with TENSORcnt.
// clang-23 toolchain: 6-arg builtin (extra int32x8 group before cpol).
// ---------------------------------------------------------------------------
__device__ __forceinline__ void tdm_load_2d(unsigned lds_off, unsigned long long gaddr,
                                            unsigned tdim0, unsigned tdim1,
                                            unsigned long long stride0,
                                            unsigned tile0, unsigned tile1,
                                            unsigned padIntervalCode, unsigned padAmountCode)
{
  u32x4 g0;
  g0[0] = 1u;                                   // count=1, user D#
  g0[1] = lds_off;                              // lds_addr
  g0[2] = (unsigned)gaddr;                      // global_addr[31:0]
  g0[3] = (unsigned)((gaddr >> 32) & 0x01FFFFFFu) | (2u << 30);  // [56:32] | type=2
  i32x8 g1;
  g1[0] = (int)((1u << 16)                      // data_size = 2 bytes
              | (1u << 20)                      // pad_enable
              | (padIntervalCode << 22)
              | (padAmountCode << 25));
  g1[1] = (int)((tdim0 & 0xFFFFu) << 16);       // tensor_dim0[15:0]
  g1[2] = (int)((tdim0 >> 16) | ((tdim1 & 0xFFFFu) << 16));
  g1[3] = (int)((tdim1 >> 16) | (tile0 << 16)); // tile_dim0
  g1[4] = (int)tile1;                           // tile_dim1 (tile_dim2 = 0)
  g1[5] = (int)(unsigned)stride0;               // tensor_dim0_stride[31:0]
  g1[6] = (int)((unsigned)(stride0 >> 32) & 0xFFFFu);
  g1[7] = 0;
  i32x4 g2 = {0, 0, 0, 0};
  i32x4 g3 = {0, 0, 0, 0};
  i32x8 g4 = {0, 0, 0, 0, 0, 0, 0, 0};
  __builtin_amdgcn_tensor_load_to_lds(g0, g1, g2, g3, g4, 0);
}

// ---------------------------------------------------------------------------
// 16x32 bf16 A/B fragment from LDS (row-major, rowStride halves, mult of 8).
// lane L -> row = L%16; halves 0..7 -> K = kb..kb+7, halves 8..15 ->
// K = kb+16..kb+23, kb = (L>=16) ? 8 : 0.
// ---------------------------------------------------------------------------
__device__ __forceinline__ v16bf ld_frag(const __bf16* tile, int stride, int lane) {
  int r  = lane & 15;
  int kb = (lane & 16) ? 8 : 0;
  const __bf16* p = tile + r * stride + kb;
  v8bf lo = *(const v8bf*)(p);
  v8bf hi = *(const v8bf*)(p + 16);
  return __builtin_shufflevector(lo, hi, 0,1,2,3,4,5,6,7,8,9,10,11,12,13,14,15);
}

__device__ __forceinline__ v8f wmma_bf16(v16bf a, v16bf b, v8f c) {
  return __builtin_amdgcn_wmma_f32_16x16x32_bf16(false, a, false, b, (short)0, c,
                                                 false, false);
}

// ---------------------------------------------------------------------------
// x (f32) -> bf16, vectorized; n4 = elements/4
// ---------------------------------------------------------------------------
__global__ __launch_bounds__(256)
void convert_f32_to_bf16(const float* __restrict__ in, __bf16* __restrict__ out, int n4)
{
  int i = blockIdx.x * 256 + threadIdx.x;
  if (i < n4) {
    v4f f = *(const v4f*)(in + (size_t)i * 4);
    __bf16* o = out + (size_t)i * 4;
    o[0] = (__bf16)f[0]; o[1] = (__bf16)f[1];
    o[2] = (__bf16)f[2]; o[3] = (__bf16)f[3];
  }
}

// ---------------------------------------------------------------------------
// W (K x N, f32) -> Wt (N x K, bf16): run once so GEMM B tiles are K-contiguous.
// ---------------------------------------------------------------------------
__global__ __launch_bounds__(256)
void transpose_f32_to_bf16(const float* __restrict__ W, __bf16* __restrict__ Wt,
                           int K, int N)
{
  __shared__ __bf16 tile[32][33];
  const int n0 = blockIdx.x * 32;
  const int k0 = blockIdx.y * 32;
  const int c  = threadIdx.x & 31;
  const int r0 = threadIdx.x >> 5;
  for (int r = r0; r < 32; r += 8)
    tile[r][c] = (__bf16)W[(size_t)(k0 + r) * N + n0 + c];
  __syncthreads();
  for (int r = r0; r < 32; r += 8)
    Wt[(size_t)(n0 + r) * K + k0 + c] = tile[c][r];
}

// ---------------------------------------------------------------------------
// C(M x N) = A(M x K) * Bt(N x K)^T, all bf16 in, f32 accumulate.
// Block tile 128x256, 256 threads = 8 waves (2x4), wave tile 64x64 = 16 WMMA
// fragments. Tiles staged by the TDM (wave 0 issues, TENSORcnt + barrier
// joins), double-buffered so DMA overlaps WMMA. N, K compile-time so the
// epilogue folds row steps into store immediates (no accumulator spills).
// ---------------------------------------------------------------------------
template <int N, int K, bool F32OUT>
__global__ __launch_bounds__(256, 2)
void gemm_bf16_tdm(const __bf16* __restrict__ A, const __bf16* __restrict__ BT,
                   __bf16* __restrict__ Cb, float* __restrict__ Cf,
                   const float* __restrict__ bias)
{
  __shared__ __bf16 As[2][128 * 40];   // 128 rows x 32 k, stride 40 (TDM pad)
  __shared__ __bf16 Bs[2][256 * 40];   // 256 rows x 32 k, stride 40

  const int tid  = threadIdx.x;
  const int lane = tid & 31;
  const int wave = tid >> 5;
  const int wm   = wave >> 2;          // 0..1 -> 64 rows
  const int wn   = wave & 3;           // 0..3 -> 64 cols
  const int n0   = blockIdx.x * 256;
  const int m0   = blockIdx.y * 128;

  const unsigned long long aBase =
      (unsigned long long)(uintptr_t)A + (size_t)m0 * K * 2;
  const unsigned long long bBase =
      (unsigned long long)(uintptr_t)BT + (size_t)n0 * K * 2;

  v8f z = {};
  v8f acc[4][4];
  for (int i = 0; i < 4; ++i)
    for (int j = 0; j < 4; ++j) acc[i][j] = z;

  constexpr int nkt = K >> 5;
  if (wave == 0) {
    // pad codes: interval 3 -> 16 dwords (64B row), amount 3 -> 4 dwords (16B)
    tdm_load_2d((unsigned)(uintptr_t)&As[0][0], aBase, K, 128, K, 32, 128, 3, 3);
    tdm_load_2d((unsigned)(uintptr_t)&Bs[0][0], bBase, K, 256, K, 32, 256, 3, 3);
    __builtin_amdgcn_s_wait_tensorcnt(0);
  }
  __syncthreads();

  for (int kt = 0; kt < nkt; ++kt) {
    const int buf = kt & 1;
    if (wave == 0 && kt + 1 < nkt) {       // prefetch next tiles into other buffer
      tdm_load_2d((unsigned)(uintptr_t)&As[buf ^ 1][0], aBase + (size_t)(kt + 1) * 64,
                  K, 128, K, 32, 128, 3, 3);
      tdm_load_2d((unsigned)(uintptr_t)&Bs[buf ^ 1][0], bBase + (size_t)(kt + 1) * 64,
                  K, 256, K, 32, 256, 3, 3);
    }
    v16bf af[4], bfr[4];
    for (int fm = 0; fm < 4; ++fm)
      af[fm] = ld_frag(&As[buf][(wm * 64 + fm * 16) * 40], 40, lane);
    for (int fn = 0; fn < 4; ++fn)
      bfr[fn] = ld_frag(&Bs[buf][(wn * 64 + fn * 16) * 40], 40, lane);
    for (int fm = 0; fm < 4; ++fm)
      for (int fn = 0; fn < 4; ++fn)
        acc[fm][fn] = wmma_bf16(af[fm], bfr[fn], acc[fm][fn]);
    if (kt + 1 < nkt) {
      if (wave == 0) __builtin_amdgcn_s_wait_tensorcnt(0);
      __syncthreads();
    }
  }

  // ---- epilogue: one base per fragment, row steps are compile-time (N) ----
  const int hi8 = (lane & 16) ? 8 : 0;
  const int nl  = lane & 15;
  const int col0 = n0 + wn * 64 + nl;
  const size_t rbase = (size_t)(m0 + wm * 64 + hi8) * N + col0;
  for (int fm = 0; fm < 4; ++fm) {
    for (int fn = 0; fn < 4; ++fn) {
      if constexpr (F32OUT) {
        float bb = bias[col0 + fn * 16];
        float* p = Cf + rbase + (size_t)fm * 16 * N + fn * 16;
        for (int r = 0; r < 8; ++r)
          p[(size_t)r * N] = acc[fm][fn][r] + bb;
      } else {
        __bf16* p = Cb + rbase + (size_t)fm * 16 * N + fn * 16;
        for (int r = 0; r < 8; ++r)
          p[(size_t)r * N] = (__bf16)acc[fm][fn][r];
      }
    }
  }
}

// ---------------------------------------------------------------------------
// One workgroup per (b, h, t): 196x196 spatial attention (hd=64); temporal
// branch reduces to adding mean_n(v) (softmax over a length-1 axis == 1).
// Q/K tiles staged via TDM (stride 2304 -> padded stride-72 LDS rows);
// V transposed manually. Dynamic LDS: 279,552 B.
//   [0      ) S   : 208 x 228 f32  (P bf16 overlaid in place, stride 456h)
//   [189696 ) Qs  : 208 x 72  bf16
//   [219648 ) Ks  : 208 x 72  bf16
//   [249600 ) Vt  : 64  x 232 bf16
//   [279296 ) vmean: 64 f32
// ---------------------------------------------------------------------------
__global__ __launch_bounds__(256)
void attn_kernel(const __bf16* __restrict__ qkv, __bf16* __restrict__ outb)
{
  extern __shared__ char smem[];
  float*  Ssm   = (float*) (smem);
  __bf16* Psm   = (__bf16*)(smem);
  __bf16* Qs    = (__bf16*)(smem + 189696);
  __bf16* Ks    = (__bf16*)(smem + 219648);
  __bf16* Vt    = (__bf16*)(smem + 249600);
  float*  vmean = (float*) (smem + 279296);

  const int tid  = threadIdx.x;
  const int lane = tid & 31;
  const int wave = tid >> 5;

  const int blk = blockIdx.x;                 // b*96 + h*8 + t
  const int b = blk / (H_ * T_);
  const int h = (blk / T_) % H_;
  const int t = blk % T_;

  const size_t rowbase = (size_t)(b * TN + t * NS) * QKN + h * HD;

  // ---- TDM: Q and K 196x64 tiles, stride 2304, pad rows to 72 halves ----
  if (wave == 0) {
    unsigned long long qa = (unsigned long long)(uintptr_t)qkv + rowbase * 2;
    // pad codes: interval 4 -> 32 dwords (128B row), amount 3 -> 4 dwords
    tdm_load_2d((unsigned)(uintptr_t)Qs, qa,            QKN, NS, QKN, HD, NS, 4, 3);
    tdm_load_2d((unsigned)(uintptr_t)Ks, qa + D_ * 2,   QKN, NS, QKN, HD, NS, 4, 3);
  }
  // ---- manual: V transposed + zero padding (overlaps the DMA) ----
  for (int c = tid; c < NS * HD; c += 256) {          // Vt[d][n] = v[n][d]
    int n = c >> 6, d = c & 63;
    Vt[d * 232 + n] = qkv[rowbase + (size_t)n * QKN + 2 * D_ + d];
  }
  for (int c = tid; c < 12 * 72; c += 256) {          // zero Q/K pad rows 196..207
    Qs[196 * 72 + c] = (__bf16)0.f;
    Ks[196 * 72 + c] = (__bf16)0.f;
  }
  for (int c = tid; c < 64 * 36; c += 256) {          // zero Vt cols 196..231
    int d = c / 36, n = 196 + c % 36;
    Vt[d * 232 + n] = (__bf16)0.f;
  }
  if (wave == 0) __builtin_amdgcn_s_wait_tensorcnt(0);
  __syncthreads();

  // ---- S = Q K^T : 13x13 tiles, K = 64 ----
  for (int tile = wave; tile < 13 * 13; tile += 8) {
    int tm = tile / 13, tn = tile % 13;
    v8f acc = {};
    for (int ks = 0; ks < 64; ks += 32) {
      v16bf a  = ld_frag(Qs + tm * 16 * 72 + ks, 72, lane);
      v16bf bb = ld_frag(Ks + tn * 16 * 72 + ks, 72, lane);
      acc = wmma_bf16(a, bb, acc);
    }
    int hi8 = (lane & 16) ? 8 : 0;
    int nl  = lane & 15;
    for (int r = 0; r < 8; ++r)
      Ssm[(tm * 16 + r + hi8) * 228 + tn * 16 + nl] = acc[r];
  }
  __syncthreads();

  // ---- temporal branch: x_t[d] = mean_n v[n][d] ----
  if (tid < 64) {
    float s = 0.f;
    for (int n = 0; n < NS; ++n) s += (float)Vt[tid * 232 + n];
    vmean[tid] = s * (1.f / 196.f);
  }

  // ---- row softmax (scale folded into exp); P bf16 written in place ----
  const float scale = 0.125f;                  // hd^-0.5
  for (int row = wave; row < NS; row += 8) {
    const float* Sr = Ssm + row * 228;
    __bf16* Pr = Psm + row * 456;
    float sv[7], e[7];
    float lmax = -3.0e38f;
    for (int i = 0; i < 7; ++i) {
      int m = lane + 32 * i;                   // covers 0..223 exactly
      float v = (m < NS) ? Sr[m] : -3.0e38f;
      sv[i] = v;
      lmax = fmaxf(lmax, v);
    }
    for (int off = 16; off > 0; off >>= 1) lmax = fmaxf(lmax, __shfl_xor(lmax, off, 32));
    float lsum = 0.f;
    for (int i = 0; i < 7; ++i) {
      int m = lane + 32 * i;
      float v = (m < NS) ? __expf((sv[i] - lmax) * scale) : 0.f;
      e[i] = v; lsum += v;
    }
    for (int off = 16; off > 0; off >>= 1) lsum += __shfl_xor(lsum, off, 32);
    float inv = 1.f / lsum;
    for (int i = 0; i < 7; ++i) Pr[lane + 32 * i] = (__bf16)(e[i] * inv);
  }
  __syncthreads();

  // ---- O = P @ V + x_t : 13x4 tiles, K = 224 (zero-padded) ----
  for (int tile = wave; tile < 13 * 4; tile += 8) {
    int tm = tile >> 2, td = tile & 3;
    v8f acc = {};
    for (int ks = 0; ks < 224; ks += 32) {
      v16bf a  = ld_frag(Psm + tm * 16 * 456 + ks, 456, lane);
      v16bf bb = ld_frag(Vt  + td * 16 * 232 + ks, 232, lane);
      acc = wmma_bf16(a, bb, acc);
    }
    int hi8 = (lane & 16) ? 8 : 0;
    int nl  = lane & 15;
    float vm = vmean[td * 16 + nl];
    for (int r = 0; r < 8; ++r) {
      int m = tm * 16 + r + hi8;
      if (m < NS) {
        size_t orow = (size_t)(b * TN + t * NS + m);
        outb[orow * D_ + h * HD + td * 16 + nl] = (__bf16)(acc[r] + vm);
      }
    }
  }
}

// ---------------------------------------------------------------------------
extern "C" void kernel_launch(void* const* d_in, const int* in_sizes, int n_in,
                              void* d_out, int out_size, void* d_ws, size_t ws_size,
                              hipStream_t stream) {
  (void)in_sizes; (void)n_in; (void)out_size; (void)ws_size;
  const float* x      = (const float*)d_in[0];
  // d_in[1] = T, d_in[2] = N (python scalars; shapes hardcoded to match)
  const float* w_qkv  = (const float*)d_in[3];
  const float* w_proj = (const float*)d_in[4];
  const float* b_proj = (const float*)d_in[5];
  float* out = (float*)d_out;

  char* ws = (char*)d_ws;
  __bf16* qkvb   = (__bf16*)(ws);                              // 57,802,752 B
  __bf16* attnb  = (__bf16*)(ws + 57802752);                   // 19,267,584 B
  __bf16* xb     = (__bf16*)(ws + 77070336);                   // 19,267,584 B
  __bf16* wqkvT  = (__bf16*)(ws + 96337920);                   //  3,538,944 B
  __bf16* wprojT = (__bf16*)(ws + 99876864);                   //  1,179,648 B

  (void)hipFuncSetAttribute((const void*)attn_kernel,
                            hipFuncAttributeMaxDynamicSharedMemorySize, 279552);

  dim3 blk(256);
  // one-time operand prep: x -> bf16; weights -> bf16 transposed (N x K)
  convert_f32_to_bf16<<<dim3((M_ * D_ / 4 + 255) / 256), blk, 0, stream>>>(
      x, xb, M_ * D_ / 4);
  transpose_f32_to_bf16<<<dim3(QKN / 32, D_ / 32), blk, 0, stream>>>(
      w_qkv, wqkvT, D_, QKN);
  transpose_f32_to_bf16<<<dim3(D_ / 32, D_ / 32), blk, 0, stream>>>(
      w_proj, wprojT, D_, D_);
  // qkv = x @ w_qkv   (12544 x 768 x 2304), bf16 out
  gemm_bf16_tdm<QKN, D_, false><<<dim3(QKN / 256, M_ / 128), blk, 0, stream>>>(
      xb, wqkvT, qkvb, nullptr, nullptr);
  // divided space-time attention, bf16 out
  attn_kernel<<<dim3(B_ * H_ * T_), blk, 279552, stream>>>(qkvb, attnb);
  // out = attn @ w_proj + b_proj   (12544 x 768 x 768), f32 out
  gemm_bf16_tdm<D_, D_, true><<<dim3(D_ / 256, M_ / 128), blk, 0, stream>>>(
      attnb, wprojT, nullptr, out, b_proj);
}
